// SAM__40973988004733
// MI455X (gfx1250) — matrix operand order repaired and checked by reading, not compiled, not run on hardware
//
#include <hip/hip_runtime.h>
#include <hip/hip_bf16.h>
#include <stdint.h>

typedef __attribute__((ext_vector_type(16))) _Float16 v16h;
typedef __attribute__((ext_vector_type(8)))  float    v8f;

#define B_    8
#define NS_   16384
#define D_    256
#define K_    4096
#define NBOT_ (NS_ - K_)            // 12288
#define NTOK_ (B_ * NS_)            // 131072
#define OUT_ELEMS (B_ * K_ * D_)    // 8388608
#define NCOMB_BLOCKS (B_ * K_ / 8)  // 4096

// workspace layout (bytes)
#define WS_S_OFF      0
#define WS_ORDER_OFF  (NTOK_ * 4)
#define WS_PERM_OFF   (WS_ORDER_OFF + NTOK_ * 4)
#define WS_PART_OFF   (WS_PERM_OFF + B_ * K_ * 4)

// ---------------------------------------------------------------------------
// Kernel 1: fused LayerNorm-stats + rank-1 projection via WMMA.
// Per wave: 16 tokens. A = x tile (16x32 f16), B col0 = ones (-> sum x),
// col1 = gamma*w (-> dot). Second WMMA with A=x^2 gives sum x^2.
// score = (rstd*(Sxw - mu*Cgw) + Cbw + b)/sqrt(2) + gumbel + (1-drop)*-1e9
// ---------------------------------------------------------------------------
__launch_bounds__(128)
__global__ void score_kernel(const float* __restrict__ x,
                             const float* __restrict__ gamma,
                             const float* __restrict__ beta,
                             const float* __restrict__ w,
                             const float* __restrict__ bias,
                             const float* __restrict__ gumbel,
                             const float* __restrict__ drop,
                             float* __restrict__ s_out) {
  __shared__ _Float16 gw_h[D_];
  __shared__ float red[128];
  __shared__ float stS1[64], stS2[64], stSxw[64];
  __shared__ float cgw_s, cbw_s;

  const int tid = threadIdx.x;

  // per-block scalar constants Cgw = sum(gamma*w), Cbw = sum(beta*w)
  float pg = 0.f, pb = 0.f;
  for (int d = tid; d < D_; d += 128) {
    float wv = w[d];
    float g  = gamma[d] * wv;
    gw_h[d]  = (_Float16)g;
    pg += g;
    pb += beta[d] * wv;
  }
  red[tid] = pg; __syncthreads();
  for (int off = 64; off > 0; off >>= 1) {
    if (tid < off) red[tid] += red[tid + off];
    __syncthreads();
  }
  if (tid == 0) cgw_s = red[0];
  __syncthreads();
  red[tid] = pb; __syncthreads();
  for (int off = 64; off > 0; off >>= 1) {
    if (tid < off) red[tid] += red[tid + off];
    __syncthreads();
  }
  if (tid == 0) cbw_s = red[0];
  __syncthreads();

  const int wave = tid >> 5;
  const int lane = tid & 31;
  const int n    = lane & 15;   // A row / D column index
  const int hi   = lane >> 4;   // lane half selects K sub-range
  const long long tok0 = (long long)blockIdx.x * 64 + (long long)wave * 16;
  const float* xrow = x + (tok0 + n) * D_;

  v8f acc0 = {0.f,0.f,0.f,0.f,0.f,0.f,0.f,0.f};  // col0: S1, col1: Sxw
  v8f acc1 = {0.f,0.f,0.f,0.f,0.f,0.f,0.f,0.f};  // col0: S2

#pragma unroll
  for (int it = 0; it < 8; ++it) {
    const int k0   = it * 32;
    const int offA = k0 + hi * 8;     // lanes 0-15: K 0..7,16..23 ; 16-31: +8
    float4 f0 = *(const float4*)(xrow + offA);
    float4 f1 = *(const float4*)(xrow + offA + 4);
    float4 f2 = *(const float4*)(xrow + offA + 16);
    float4 f3 = *(const float4*)(xrow + offA + 20);
    float fa[16] = {f0.x,f0.y,f0.z,f0.w, f1.x,f1.y,f1.z,f1.w,
                    f2.x,f2.y,f2.z,f2.w, f3.x,f3.y,f3.z,f3.w};
    v16h a, a2;
#pragma unroll
    for (int e = 0; e < 16; ++e) {
      a[e]  = (_Float16)fa[e];
      a2[e] = (_Float16)(fa[e] * fa[e]);
    }
    const int kb = k0 + hi * 8;  // B rows held by this lane half (same K order as A)
    v16h bm;
#pragma unroll
    for (int e = 0; e < 16; ++e) {
      _Float16 g = gw_h[kb + (e < 8 ? e : e + 8)];
      bm[e] = (n == 0) ? (_Float16)1.0f : ((n == 1) ? g : (_Float16)0.0f);
    }
    acc0 = __builtin_amdgcn_wmma_f32_16x16x32_f16(false, a,  false, bm, (short)0, acc0, false, false);
    acc1 = __builtin_amdgcn_wmma_f32_16x16x32_f16(false, a2, false, bm, (short)0, acc1, false, false);
  }

  // D layout: VGPR j, lanes 0-15 -> M=j ; lanes 16-31 -> M=j+8 ; N = lane&15
  // Uniform store blocks (select destination pointer) to avoid per-store exec
  // mask juggling.
  if (n < 2) {
    float* dst = (n == 0) ? stS1 : stSxw;
    const int tl0 = wave * 16 + hi * 8;
#pragma unroll
    for (int j = 0; j < 8; ++j) dst[tl0 + j] = acc0[j];
  }
  if (n == 0) {
    const int tl0 = wave * 16 + hi * 8;
#pragma unroll
    for (int j = 0; j < 8; ++j) stS2[tl0 + j] = acc1[j];
  }
  __syncthreads();

  if (tid < 64) {
    long long tf = (long long)blockIdx.x * 64 + tid;
    float S1 = stS1[tid], S2 = stS2[tid], Sxw = stSxw[tid];
    float mu   = S1 * (1.0f / D_);
    float var  = S2 * (1.0f / D_) - mu * mu;
    float rstd = rsqrtf(var + 1e-5f);
    float raw  = rstd * (Sxw - mu * cgw_s) + cbw_s + bias[0];
    float sc   = raw * 0.70710678118654752f
               + gumbel[tf] + (1.0f - drop[tf]) * (-1e9f);
    s_out[tf] = sc;
  }
}

// ---------------------------------------------------------------------------
// Kernel 2: per-batch bitonic argsort in LDS (16384 u64 keys = 128KB).
// Keys staged into LDS with CDNA5 async loads (global_load_async_to_lds_b128,
// s_wait_asynccnt). Packed key = (monotone(f) [maybe inverted] << 32) | idx,
// sorted ascending -> stable tie-break on index like jnp.argsort.
// ---------------------------------------------------------------------------
__launch_bounds__(1024)
__global__ void sort_kernel(const float* __restrict__ keys, int key_stride,
                            int n_valid, int descending,
                            int* __restrict__ out_idx, int out_stride, int n_out) {
  extern __shared__ unsigned char smem[];
  unsigned long long* packed = (unsigned long long*)smem;            // 128 KB
  float* raw = (float*)(smem + (size_t)NS_ * 8);                     // 64 KB
  const int tid = threadIdx.x;
  const float* base = keys + (long long)blockIdx.x * key_stride;

  // async stage raw keys into LDS (GVS mode: sgpr base + per-lane voffset)
  {
    const int nb = n_valid >> 2;  // 16-byte chunks
    unsigned rawOff = (unsigned)(uintptr_t)(void*)raw;
    for (int c = tid; c < nb; c += 1024) {
      unsigned voff  = (unsigned)(c * 16);
      unsigned laddr = rawOff + voff;
      asm volatile("global_load_async_to_lds_b128 %0, %1, %2 offset:0"
                   :: "v"(laddr), "v"(voff), "s"(base) : "memory");
    }
    asm volatile("s_wait_asynccnt 0" ::: "memory");
  }
  __syncthreads();

  // pack keys (pad region sorts to the end)
  for (int i = tid; i < NS_; i += 1024) {
    unsigned long long kv;
    if (i < n_valid) {
      unsigned u = __float_as_uint(raw[i]);
      u = (u & 0x80000000u) ? ~u : (u | 0x80000000u);  // monotone float->uint
      if (descending) u = ~u;
      kv = ((unsigned long long)u << 32) | (unsigned)i;
    } else {
      kv = 0xFFFFFFFFFFFFFFFFull;
    }
    packed[i] = kv;
  }
  __syncthreads();

  // bitonic sort, ascending
  for (int k = 2; k <= NS_; k <<= 1) {
    for (int j = k >> 1; j > 0; j >>= 1) {
      for (int t = tid; t < (NS_ >> 1); t += 1024) {
        int i = ((t & ~(j - 1)) << 1) | (t & (j - 1));
        int p = i | j;
        bool up = ((i & k) == 0);
        unsigned long long a = packed[i];
        unsigned long long c = packed[p];
        if ((a > c) == up) { packed[i] = c; packed[p] = a; }
      }
      __syncthreads();
    }
  }

  for (int i = tid; i < n_out; i += 1024)
    out_idx[(long long)blockIdx.x * out_stride + i] = (int)(packed[i] & 0xFFFFFFFFu);
}

// ---------------------------------------------------------------------------
// Kernel 3: gather + softplus blend. One wave per output row (256 floats).
// ---------------------------------------------------------------------------
__launch_bounds__(256)
__global__ void combine_kernel(const float* __restrict__ x,
                               const float* __restrict__ s,
                               const int* __restrict__ order,
                               const int* __restrict__ perm,
                               float* __restrict__ out,
                               float* __restrict__ partials) {
  __shared__ float wsum[8];
  const int tid  = threadIdx.x;
  const int wave = tid >> 5;
  const int lane = tid & 31;
  const long long r = (long long)blockIdx.x * 8 + wave;  // row in [0, B*K)
  const int b = (int)(r / K_);
  const int i = (int)(r % K_);

  const int ti = order[(long long)b * NS_ + i];
  const int p  = perm[r];
  const int bi = order[(long long)b * NS_ + K_ + p];
  const float ts = s[(long long)b * NS_ + ti];
  const float bs = s[(long long)b * NS_ + bi];
  float x1  = fmaxf(ts, 0.f) + log1pf(expf(-fabsf(ts)));  // softplus
  float x2  = fmaxf(bs, 0.f) + log1pf(expf(-fabsf(bs)));
  float inv = 1.0f / (x1 + x2 + 1e-9f);
  float a1 = x1 * inv, a2 = x2 * inv;

  const float* tr = x + ((long long)b * NS_ + ti) * D_;
  const float* br = x + ((long long)b * NS_ + bi) * D_;
  float* orow = out + r * D_;
  const int d0 = lane * 8;
  float4 t0 = *(const float4*)(tr + d0);
  float4 t1 = *(const float4*)(tr + d0 + 4);
  float4 u0 = *(const float4*)(br + d0);
  float4 u1 = *(const float4*)(br + d0 + 4);
  float4 o0, o1;
  o0.x = t0.x * a1 + u0.x * a2;  o0.y = t0.y * a1 + u0.y * a2;
  o0.z = t0.z * a1 + u0.z * a2;  o0.w = t0.w * a1 + u0.w * a2;
  o1.x = t1.x * a1 + u1.x * a2;  o1.y = t1.y * a1 + u1.y * a2;
  o1.z = t1.z * a1 + u1.z * a2;  o1.w = t1.w * a1 + u1.w * a2;
  *(float4*)(orow + d0)     = o0;
  *(float4*)(orow + d0 + 4) = o1;

  if (lane == 0) wsum[wave] = a2;
  __syncthreads();
  if (tid == 0) {
    float acc = 0.f;
    for (int wv = 0; wv < 8; ++wv) acc += wsum[wv];
    partials[blockIdx.x] = acc;
  }
}

// ---------------------------------------------------------------------------
// Kernel 4: deterministic final reduction -> reg = mean(x2/xs)
// ---------------------------------------------------------------------------
__launch_bounds__(256)
__global__ void reduce_kernel(const float* __restrict__ partials,
                              float* __restrict__ reg_out) {
  __shared__ float red[256];
  float acc = 0.f;
  for (int i = threadIdx.x; i < NCOMB_BLOCKS; i += 256) acc += partials[i];
  red[threadIdx.x] = acc;
  __syncthreads();
  for (int off = 128; off > 0; off >>= 1) {
    if (threadIdx.x < off) red[threadIdx.x] += red[threadIdx.x + off];
    __syncthreads();
  }
  if (threadIdx.x == 0) reg_out[0] = red[0] * (1.0f / (float)(B_ * K_));
}

extern "C" void kernel_launch(void* const* d_in, const int* in_sizes, int n_in,
                              void* d_out, int out_size, void* d_ws, size_t ws_size,
                              hipStream_t stream) {
  const float* x      = (const float*)d_in[0];
  const float* gamma  = (const float*)d_in[1];
  const float* beta   = (const float*)d_in[2];
  const float* w      = (const float*)d_in[3];
  const float* bias   = (const float*)d_in[4];
  const float* gumbel = (const float*)d_in[5];
  const float* drop   = (const float*)d_in[6];
  const float* noise  = (const float*)d_in[7];
  float* out = (float*)d_out;

  unsigned char* ws = (unsigned char*)d_ws;
  float* s        = (float*)(ws + WS_S_OFF);
  int*   order    = (int*)(ws + WS_ORDER_OFF);
  int*   perm     = (int*)(ws + WS_PERM_OFF);
  float* partials = (float*)(ws + WS_PART_OFF);

  score_kernel<<<NTOK_ / 64, 128, 0, stream>>>(x, gamma, beta, w, bias,
                                               gumbel, drop, s);

  const size_t smem = (size_t)NS_ * 8 + (size_t)NS_ * 4;  // 192 KB (<320KB/WGP)
  // full descending argsort of scores, per batch
  sort_kernel<<<B_, 1024, smem, stream>>>(s, NS_, NS_, 1, order, NS_, NS_);
  // ascending argsort of shuffle noise (12288 valid, keep first 4096)
  sort_kernel<<<B_, 1024, smem, stream>>>(noise, NBOT_, NBOT_, 0, perm, K_, K_);

  combine_kernel<<<NCOMB_BLOCKS, 256, 0, stream>>>(x, s, order, perm, out, partials);
  reduce_kernel<<<1, 256, 0, stream>>>(partials, out + OUT_ELEMS);
}
